// Decoder_38783554682995
// MI455X (gfx1250) — matrix operand order repaired and checked by reading, not compiled, not run on hardware
//
#include <hip/hip_runtime.h>
#include <hip/hip_fp16.h>
#include <hip/hip_bf16.h>

typedef _Float16 h16;
typedef __attribute__((ext_vector_type(16))) _Float16 v16h;
typedef __attribute__((ext_vector_type(8)))  float    v8f;

#define LOG_EPS (-11.512925464970229f)

// ---------------------------------------------------------------------------
// WMMA fragment loaders (CDNA5 16x16x32 f16, wave32 layouts per ISA 7.12.2)
// ---------------------------------------------------------------------------
union AFrag { v16h v; uint4 u[2]; };

// A matrix 16x32 f16, row-major with leading dim lda (halves).
// lanes 0-15: row M=lane,   K = {k0..k0+7} U {k0+16..k0+23}
// lanes16-31: row M=lane-16,K = {k0+8..k0+15} U {k0+24..k0+31}
__device__ __forceinline__ v16h load_a_frag(const h16* A, int lda, int m0, int k0, int lane) {
    int m   = m0 + (lane & 15);
    int off = (lane >= 16) ? 8 : 0;
    const h16* base = A + (long)m * lda + k0 + off;
    AFrag f;
    f.u[0] = *(const uint4*)(base);        // K off..off+7
    f.u[1] = *(const uint4*)(base + 16);   // K off+16..off+23
    return f.v;
}

// B matrix 32x16 f16 where B[k][n] = W[n][k], W row-major [N,K] (weight form).
// lane holds column n = lane&15, K = {k0..k0+7} U {k0+16..k0+23} (lane>=16 -> +8).
__device__ __forceinline__ v16h load_b_wt(const h16* W, int ldw, int n0, int k0, int lane) {
    int n  = n0 + (lane & 15);
    int kb = k0 + ((lane >= 16) ? 8 : 0);
    const h16* base = W + (long)n * ldw + kb;
    AFrag f;
    f.u[0] = *(const uint4*)(base);
    f.u[1] = *(const uint4*)(base + 16);
    return f.v;
}

// ---------------------------------------------------------------------------
// Strided-batched WMMA GEMM: C = act(A[M,K] * W[N,K]^T + bias[N])
// REQUIRES: M % 16 == 0, N % 64 == 0, K % 32 == 0.
// Each wave owns a 16x64 output strip: one A fragment feeds 4 unconditional
// back-to-back WMMAs (accumulators stay pinned, EXEC stays all-ones).
// ---------------------------------------------------------------------------
__global__ void __launch_bounds__(256) gemm16(
    const h16* __restrict__ A, long sA, int lda,
    const h16* __restrict__ W, long sB, int ldw,
    const float* __restrict__ bias,
    float* __restrict__ Cf, long sCf, int ldcf,
    h16* __restrict__ Ch, long sCh, int ldch,
    int M, int N, int K, int relu)
{
    int lane = threadIdx.x & 31;
    int wave = threadIdx.x >> 5;
    int ng  = N >> 6;                 // groups of 64 columns
    long tile = (long)blockIdx.x * 8 + wave;
    long total = (long)(M >> 4) * ng;
    if (tile >= total) return;
    int mt = (int)(tile / ng), g = (int)(tile % ng);
    int m0 = mt << 4;
    int nbase = g << 6;
    int bz = blockIdx.y;
    const h16* Ab = A + (long)bz * sA;
    const h16* Wb = W + (long)bz * sB;
    int colL = lane & 15;

    v8f acc0, acc1, acc2, acc3;
    {
        float b0 = bias ? bias[nbase +  0 + colL] : 0.0f;
        float b1 = bias ? bias[nbase + 16 + colL] : 0.0f;
        float b2 = bias ? bias[nbase + 32 + colL] : 0.0f;
        float b3 = bias ? bias[nbase + 48 + colL] : 0.0f;
#pragma unroll
        for (int i = 0; i < 8; ++i) { acc0[i] = b0; acc1[i] = b1; acc2[i] = b2; acc3[i] = b3; }
    }

    int arow = m0 + colL;
    for (int k0 = 0; k0 < K; k0 += 32) {
        v16h a  = load_a_frag(Ab, lda, m0, k0, lane);
        v16h b0 = load_b_wt(Wb, ldw, nbase +  0, k0, lane);
        v16h b1 = load_b_wt(Wb, ldw, nbase + 16, k0, lane);
        v16h b2 = load_b_wt(Wb, ldw, nbase + 32, k0, lane);
        v16h b3 = load_b_wt(Wb, ldw, nbase + 48, k0, lane);
        // speculative prefetch of the A stream two K-tiles ahead
        __builtin_prefetch((const void*)(Ab + (long)arow * lda + k0 + 64), 0, 1);
        acc0 = __builtin_amdgcn_wmma_f32_16x16x32_f16(false, a, false, b0, (short)0, acc0, false, false);
        acc1 = __builtin_amdgcn_wmma_f32_16x16x32_f16(false, a, false, b1, (short)0, acc1, false, false);
        acc2 = __builtin_amdgcn_wmma_f32_16x16x32_f16(false, a, false, b2, (short)0, acc2, false, false);
        acc3 = __builtin_amdgcn_wmma_f32_16x16x32_f16(false, a, false, b3, (short)0, acc3, false, false);
    }

    int mrow = m0 + ((lane >= 16) ? 8 : 0);
    v8f* accs[4] = { &acc0, &acc1, &acc2, &acc3 };
#pragma unroll
    for (int j = 0; j < 4; ++j) {
        int col = nbase + (j << 4) + colL;
#pragma unroll
        for (int v = 0; v < 8; ++v) {
            float x = (*accs[j])[v];
            if (relu) x = fmaxf(x, 0.0f);
            long r = (long)(mrow + v);
            if (Cf) Cf[(long)bz * sCf + r * ldcf + col] = x;
            if (Ch) Ch[(long)bz * sCh + r * ldch + col] = (h16)x;
        }
    }
}

// ---------------------------------------------------------------------------
// Persistent sequential GRU over T steps. Fixed B=16, C=1024, 3 gate blocks.
// 8 blocks x 8 waves = 64 waves; wave g owns hidden cols [16g,16g+16).
// h ping-pong in global (f16), staged to LDS each step (ds_load_b128 A-frags).
// Grid-wide sync via monotonic atomic counter + s_sleep spin.
// ---------------------------------------------------------------------------
#define GB 16
#define GC 1024
__global__ void __launch_bounds__(256) gru_seq(
    const float* __restrict__ xw,   // [B,T,3C] input projections (+bih)
    const h16*  __restrict__ whh,   // [3C, C] f16 recurrent weights
    const float* __restrict__ bhh,  // [3C]
    h16*  __restrict__ hping,       // [2, B, C] f16, zero-initialized
    float* __restrict__ xout,       // [B,T,C] output (+= residual if set)
    h16*  __restrict__ xhout,       // [B,T,C] f16 copy of output
    unsigned* __restrict__ cnt,     // zero-initialized barrier counter
    int T, int residual)
{
    __shared__ h16 hs[GB * GC];     // 32 KB of the 320 KB WGP LDS
    int lane = threadIdx.x & 31;
    int wave = threadIdx.x >> 5;
    int g  = blockIdx.x * 8 + wave; // 0..63
    int j0 = g << 4;
    int colL = lane & 15;
    float br = bhh[j0 + colL];
    float bz = bhh[GC + j0 + colL];
    float bn = bhh[2 * GC + j0 + colL];
    unsigned nb = gridDim.x;

    for (int t = 0; t < T; ++t) {
        const h16* cur = hping + (long)(t & 1) * (GB * GC);
        h16*       nxt = hping + (long)((t & 1) ^ 1) * (GB * GC);

        // stage current hidden state into LDS (whole block cooperates)
        {
            const uint4* src = (const uint4*)cur;
            uint4* dst = (uint4*)hs;
            for (int i = threadIdx.x; i < (GB * GC) / 8; i += 256) dst[i] = src[i];
        }
        __syncthreads();

        v8f ar, az, an_;
#pragma unroll
        for (int i = 0; i < 8; ++i) { ar[i] = br; az[i] = bz; an_[i] = bn; }

        for (int k0 = 0; k0 < GC; k0 += 32) {
            v16h a  = load_a_frag(hs, GC, 0, k0, lane);            // ds_load_b128
            v16h b0 = load_b_wt(whh, GC, j0,          k0, lane);   // r gate
            v16h b1 = load_b_wt(whh, GC, GC + j0,     k0, lane);   // z gate
            v16h b2 = load_b_wt(whh, GC, 2 * GC + j0, k0, lane);   // n gate
            ar  = __builtin_amdgcn_wmma_f32_16x16x32_f16(false, a, false, b0, (short)0, ar,  false, false);
            az  = __builtin_amdgcn_wmma_f32_16x16x32_f16(false, a, false, b1, (short)0, az,  false, false);
            an_ = __builtin_amdgcn_wmma_f32_16x16x32_f16(false, a, false, b2, (short)0, an_, false, false);
        }

        int col   = j0 + colL;
        int mbase = (lane >= 16) ? 8 : 0;
#pragma unroll
        for (int v = 0; v < 8; ++v) {
            int m = mbase + v;                       // batch index
            long xwb = ((long)m * T + t) * (3 * GC);
            float xr = xw[xwb + col];
            float xz = xw[xwb + GC + col];
            float xn = xw[xwb + 2 * GC + col];
            float r = 1.0f / (1.0f + __expf(-(xr + ar[v])));
            float z = 1.0f / (1.0f + __expf(-(xz + az[v])));
            float n = tanhf(xn + r * an_[v]);
            float hp = (float)hs[m * GC + col];
            float hn = (1.0f - z) * n + z * hp;
            nxt[m * GC + col] = (h16)hn;
            long oidx = ((long)m * T + t) * GC + col;
            float ov = hn + (residual ? xout[oidx] : 0.0f);
            xout[oidx]  = ov;
            xhout[oidx] = (h16)ov;
        }

        // device-wide barrier between dependent timesteps
        __syncthreads();
        if (threadIdx.x == 0) {
            __threadfence();
            __hip_atomic_fetch_add(cnt, 1u, __ATOMIC_ACQ_REL, __HIP_MEMORY_SCOPE_AGENT);
            unsigned target = nb * (unsigned)(t + 1);
            while (__hip_atomic_load(cnt, __ATOMIC_ACQUIRE, __HIP_MEMORY_SCOPE_AGENT) < target)
                __builtin_amdgcn_s_sleep(2);
            __threadfence();
        }
        __syncthreads();
    }
}

// ---------------------------------------------------------------------------
// Elementwise / data-movement helpers
// ---------------------------------------------------------------------------
__global__ void cvt_f32_h16(const float* __restrict__ in, h16* __restrict__ out,
                            long n, float scale) {
    long i = (long)blockIdx.x * 256 + threadIdx.x;
    if (i < n) out[i] = (h16)(in[i] * scale);
}

// pad rows and cols: out[rout, cout], zero outside [rin, cin]
__global__ void cvt_pad(const float* __restrict__ in, h16* __restrict__ out,
                        int rin, int cin, int rout, int cout) {
    long i = (long)blockIdx.x * 256 + threadIdx.x;
    long tot = (long)rout * cout;
    if (i >= tot) return;
    int c = (int)(i % cout);
    int r = (int)(i / cout);
    out[i] = (h16)((r < rin && c < cin) ? in[(long)r * cin + c] : 0.0f);
}

// shifted[b,t,c] = (t==0 ? LOG_EPS : gt[b,t-1,c]), zero-padded K 80 -> 96
__global__ void shift_pad(const float* __restrict__ gt, h16* __restrict__ out,
                          int B, int T, int M, int Kp) {
    long i = (long)blockIdx.x * 256 + threadIdx.x;
    long tot = (long)B * T * Kp;
    if (i >= tot) return;
    int c = (int)(i % Kp);
    long bt = i / Kp;
    int t = (int)(bt % T);
    int b = (int)(bt / T);
    float v;
    if (c >= M)      v = 0.0f;
    else if (t == 0) v = LOG_EPS;
    else             v = gt[((long)b * T + (t - 1)) * M + c];
    out[i] = (h16)v;
}

// slice columns: out[r, 0:cout] = in[r, 0:cout] with in ld = cin
__global__ void slice_cols(const float* __restrict__ in, float* __restrict__ out,
                           long rows, int cin, int cout) {
    long i = (long)blockIdx.x * 256 + threadIdx.x;
    long tot = rows * cout;
    if (i >= tot) return;
    int c = (int)(i % cout);
    long r = i / cout;
    out[i] = in[r * cin + c];
}

// LDS-tiled f16 transpose: out[b][c][r] = in[b][r][c]; rows, cols multiples of 32
__global__ void __launch_bounds__(256) transpose_h16(
    const h16* __restrict__ in, h16* __restrict__ out, int rows, int cols)
{
    __shared__ h16 tile[32][33];
    int b = blockIdx.z;
    const h16* ib = in + (long)b * rows * cols;
    h16* ob = out + (long)b * rows * cols;
    int c0 = blockIdx.x << 5, r0 = blockIdx.y << 5;
    int tx = threadIdx.x & 31, ty = threadIdx.x >> 5;   // 32x8
#pragma unroll
    for (int i = 0; i < 32; i += 8)
        tile[ty + i][tx] = ib[(long)(r0 + ty + i) * cols + (c0 + tx)];
    __syncthreads();
#pragma unroll
    for (int i = 0; i < 32; i += 8)
        ob[(long)(c0 + ty + i) * rows + (r0 + tx)] = tile[tx][ty + i];
}

// masked softmax over S=256 per (b,t) row; writes f32 alpha (in place) + f16 copy
__global__ void __launch_bounds__(256) softmax_mask(
    float* __restrict__ e, h16* __restrict__ ah,
    const float* __restrict__ mask, int T, int S)
{
    __shared__ float red[256];
    int r = blockIdx.x;
    int b = r / T;
    int s = threadIdx.x;
    long idx = (long)r * S + s;
    float v = e[idx];
    v = (mask[(long)b * S + s] > 0.0f) ? v : -1e9f;
    red[s] = v;
    __syncthreads();
    for (int off = 128; off > 0; off >>= 1) {
        if (s < off) red[s] = fmaxf(red[s], red[s + off]);
        __syncthreads();
    }
    float mx = red[0];
    __syncthreads();
    float ex = __expf(v - mx);
    red[s] = ex;
    __syncthreads();
    for (int off = 128; off > 0; off >>= 1) {
        if (s < off) red[s] += red[s + off];
        __syncthreads();
    }
    float a = ex / red[0];
    e[idx]  = a;
    ah[idx] = (h16)a;
}

// ---------------------------------------------------------------------------
// Host orchestration
// ---------------------------------------------------------------------------
extern "C" void kernel_launch(void* const* d_in, const int* in_sizes, int n_in,
                              void* d_out, int out_size, void* d_ws, size_t ws_size,
                              hipStream_t stream) {
    (void)in_sizes; (void)n_in; (void)out_size; (void)ws_size;
    const int B = 16, S = 256, T = 1024, I = 512, C = 1024, MM = 80, AA = 128, L = 2, H = 256;
    const long BT = (long)B * T;

    const float* inputs = (const float*)d_in[0];
    const float* mask   = (const float*)d_in[1];
    const float* gt     = (const float*)d_in[2];
    const float* pw1    = (const float*)d_in[3];
    const float* pb1    = (const float*)d_in[4];
    const float* pw2    = (const float*)d_in[5];
    const float* pb2    = (const float*)d_in[6];
    const float* aq     = (const float*)d_in[7];
    const float* ak     = (const float*)d_in[8];
    const float* bwih   = (const float*)d_in[9];
    const float* bwhh   = (const float*)d_in[10];
    const float* bbih   = (const float*)d_in[11];
    const float* bbhh   = (const float*)d_in[12];
    const float* gwih   = (const float*)d_in[13];
    const float* gwhh   = (const float*)d_in[14];
    const float* gbih   = (const float*)d_in[15];
    const float* gbhh   = (const float*)d_in[16];
    const float* projw  = (const float*)d_in[17];
    const float* projb  = (const float*)d_in[18];

    float* mel   = (float*)d_out;            // [B,T,80]
    float* alpha = mel + BT * MM;            // [B,T,S]

    // bump allocator over workspace
    char* wp = (char*)d_ws;
    auto alloc = [&](size_t bytes) -> void* {
        void* r = (void*)wp;
        wp += (bytes + 255) & ~(size_t)255;
        return r;
    };
    h16* shifted_h = (h16*)alloc(BT * 96 * 2);
    h16* h1_h      = (h16*)alloc(BT * 256 * 2);
    h16* xcat      = (h16*)alloc(BT * 768 * 2);     // [aligned(512) | preproc(256)]
    h16* q_h       = (h16*)alloc(BT * 128 * 2);
    h16* k_h       = (h16*)alloc((long)B * S * 128 * 2);
    h16* inputs_h  = (h16*)alloc((long)B * S * 512 * 2);
    h16* inputs_hT = (h16*)alloc((long)B * 512 * 256 * 2);   // [B, I, S]
    h16* alpha_h   = (h16*)alloc(BT * 256 * 2);
    float* xw      = (float*)alloc(BT * 3072 * 4);
    float* xbuf    = (float*)alloc(BT * 1024 * 4);
    h16* x_h       = (h16*)alloc(BT * 1024 * 2);
    float* mel_pad = (float*)alloc(BT * 128 * 4);            // padded projection out
    h16* pw1_h     = (h16*)alloc(256 * 96 * 2);
    h16* pw2_h     = (h16*)alloc(256 * 256 * 2);
    h16* aq_h      = (h16*)alloc(128 * 256 * 2);
    h16* ak_h      = (h16*)alloc(128 * 512 * 2);
    h16* bwih_h    = (h16*)alloc((long)3072 * 768 * 2);
    h16* bwhh_h    = (h16*)alloc((long)3072 * 1024 * 2);
    h16* gwih_h    = (h16*)alloc((long)L * 3072 * 1024 * 2);
    h16* gwhh_h    = (h16*)alloc((long)L * 3072 * 1024 * 2);
    h16* projw_h   = (h16*)alloc(128 * 1024 * 2);            // zero row-padded 80->128
    float* projb_p = (float*)alloc(128 * 4);
    h16* hping     = (h16*)alloc(2 * 16 * 1024 * 2);
    unsigned* cnt  = (unsigned*)alloc(256);

    auto cvt = [&](const float* in, h16* out, long n, float sc) {
        cvt_f32_h16<<<dim3((unsigned)((n + 255) / 256)), 256, 0, stream>>>(in, out, n, sc);
    };
    // f16 weight/activation conversions
    cvt(inputs, inputs_h, (long)B * S * I, 1.0f);
    cvt(pw2, pw2_h, 256 * 256, 1.0f);
    cvt(aq, aq_h, 128 * 256, 1.0f / sqrtf((float)AA));   // fold attention scale into Wq
    cvt(ak, ak_h, 128 * 512, 1.0f);
    cvt(bwih, bwih_h, (long)3072 * 768, 1.0f);
    cvt(bwhh, bwhh_h, (long)3072 * 1024, 1.0f);
    cvt(gwih, gwih_h, (long)L * 3072 * 1024, 1.0f);
    cvt(gwhh, gwhh_h, (long)L * 3072 * 1024, 1.0f);
    cvt_pad<<<dim3((256 * 96 + 255) / 256), 256, 0, stream>>>(pw1, pw1_h, 256, MM, 256, 96);
    cvt_pad<<<dim3((128 * 1024 + 255) / 256), 256, 0, stream>>>(projw, projw_h, MM, 1024, 128, 1024);
    hipMemsetAsync(projb_p, 0, 128 * 4, stream);
    hipMemcpyAsync(projb_p, projb, MM * 4, hipMemcpyDeviceToDevice, stream);
    shift_pad<<<dim3((unsigned)((BT * 96 + 255) / 256)), 256, 0, stream>>>(gt, shifted_h, B, T, MM, 96);
    // inputs^T per batch: [S,I] -> [I,S] so the aligned GEMM uses b128 B-loads
    transpose_h16<<<dim3(I / 32, S / 32, B), 256, 0, stream>>>(inputs_h, inputs_hT, S, I);

    auto gemm = [&](const h16* A, long sA, int lda,
                    const h16* W, long sB, int ldw,
                    const float* bias,
                    float* Cf, long sCf, int ldcf,
                    h16* Ch, long sCh, int ldch,
                    int Mr, int Nc, int Kd, int relu, int batch) {
        long tiles = (long)(Mr / 16) * (Nc / 64);
        dim3 grid((unsigned)((tiles + 7) / 8), (unsigned)batch);
        gemm16<<<grid, 256, 0, stream>>>(A, sA, lda, W, sB, ldw, bias,
                                         Cf, sCf, ldcf, Ch, sCh, ldch, Mr, Nc, Kd, relu);
    };

    // prenet
    gemm(shifted_h, 0, 96, pw1_h, 0, 96, pb1, nullptr, 0, 0,
         h1_h, 0, 256, (int)BT, 256, 96, 1, 1);
    gemm(h1_h, 0, 256, pw2_h, 0, 256, pb2, nullptr, 0, 0,
         xcat + 512, 0, 768, (int)BT, 256, 256, 1, 1);
    // attention projections
    gemm(xcat + 512, 0, 768, aq_h, 0, 256, nullptr, nullptr, 0, 0,
         q_h, 0, 128, (int)BT, 128, 256, 0, 1);
    gemm(inputs_h, 0, 512, ak_h, 0, 512, nullptr, nullptr, 0, 0,
         k_h, 0, 128, B * S, 128, 512, 0, 1);
    // scores e = q k^T (scale folded into aq_h), batched over B
    gemm(q_h, (long)T * 128, 128, k_h, (long)S * 128, 128, nullptr,
         alpha, (long)T * S, S, nullptr, 0, 0, T, S, 128, 0, B);
    softmax_mask<<<dim3((unsigned)BT), 256, 0, stream>>>(alpha, alpha_h, mask, T, S);
    // aligned = alpha @ inputs; B supplied as inputs^T [I,S] (weight form)
    gemm(alpha_h, (long)T * S, S, inputs_hT, (long)I * S, S, nullptr,
         nullptr, 0, 0, xcat, (long)T * 768, 768, T, I, S, 0, B);

    // blender GRU: xw = [aligned|preproc] @ bwih^T + bbih, then recurrence
    gemm(xcat, 0, 768, bwih_h, 0, 768, bbih, xw, 0, 3072,
         nullptr, 0, 0, (int)BT, 3072, 768, 0, 1);
    hipMemsetAsync(hping, 0, 2 * 16 * 1024 * 2, stream);
    hipMemsetAsync(cnt, 0, 256, stream);
    gru_seq<<<dim3(8), 256, 0, stream>>>(xw, bwhh_h, bbhh, hping, xbuf, x_h, cnt, T, 0);

    // residual GRU stack
    for (int l = 0; l < L; ++l) {
        gemm(x_h, 0, 1024, gwih_h + (long)l * 3072 * 1024, 0, 1024,
             gbih + (long)l * 3072, xw, 0, 3072, nullptr, 0, 0,
             (int)BT, 3072, 1024, 0, 1);
        hipMemsetAsync(hping, 0, 2 * 16 * 1024 * 2, stream);
        hipMemsetAsync(cnt, 0, 256, stream);
        gru_seq<<<dim3(8), 256, 0, stream>>>(xw, gwhh_h + (long)l * 3072 * 1024,
                                             gbhh + (long)l * 3072, hping,
                                             xbuf, x_h, cnt, T, 1);
    }

    // mel projection into padded scratch (N=128), then slice 80 valid cols
    gemm(x_h, 0, 1024, projw_h, 0, 1024, projb_p, mel_pad, 0, 128,
         nullptr, 0, 0, (int)BT, 128, 1024, 0, 1);
    slice_cols<<<dim3((unsigned)((BT * MM + 255) / 256)), 256, 0, stream>>>(
        mel_pad, mel, BT, 128, MM);
}
